// SGAP_38895223832724
// MI455X (gfx1250) — matrix operand order repaired and checked by reading, not compiled, not run on hardware
//
#include <hip/hip_runtime.h>
#include <hip/hip_bf16.h>

// Problem constants (from reference)
#define A_N 2048
#define F_N 128
#define H_N 128
#define B_N 512
#define L_N 64
#define G4H 512   // 4*H
#define EPSV 1e-12f

typedef __bf16 bf16_t;
typedef __attribute__((ext_vector_type(16))) __bf16 v16bf;
typedef __attribute__((ext_vector_type(8)))  float  v8f;
typedef __attribute__((ext_vector_type(2)))  float  v2f;

// ---------------------------------------------------------------------------
// Fragment helpers (wave32 WMMA; A/B k order is this code's contiguous choice, C/D per CDNA5 ISA 7.12.2)
//  A (16x32 bf16): lane L: M = L%16; elem i   -> K = (L/16)*16 + i
//  B (32x16 bf16): lane L: N = L%16; elem e   -> K = (L/16)*16 + e
//  C/D (16x16 f32): lane L: N = L%16; vgpr j -> M = (L/16)*8 + j
// ---------------------------------------------------------------------------

__device__ __forceinline__ v16bf a_frag_f32(const float* __restrict__ row,
                                            int kb, int half) {
  const float* p0 = row + kb + half * 16;
  const float* p1 = row + kb + half * 16 + 8;
  v16bf a;
#pragma unroll
  for (int r = 0; r < 8; ++r) { a[r] = (bf16_t)p0[r]; a[8 + r] = (bf16_t)p1[r]; }
  return a;
}

__device__ __forceinline__ v16bf a_frag_bf16(const bf16_t* __restrict__ row,
                                             int kb, int half) {
  const bf16_t* p0 = row + kb + half * 16;
  const bf16_t* p1 = row + kb + half * 16 + 8;
  v16bf a;
#pragma unroll
  for (int r = 0; r < 8; ++r) { a[r] = p0[r]; a[8 + r] = p1[r]; }
  return a;
}

// B from an (N x K) row-major matrix (i.e. the transposed operand): contiguous K
__device__ __forceinline__ v16bf b_frag_bf16(const bf16_t* __restrict__ w,
                                             int ldk, int n, int kb, int khalf) {
  const bf16_t* p = w + (size_t)n * ldk + kb + khalf;
  v16bf b;
#pragma unroll
  for (int e = 0; e < 16; ++e) b[e] = p[e];
  return b;
}

// B from a (K x N) row-major f32 matrix: strided K
__device__ __forceinline__ v16bf b_frag_f32_kxn(const float* __restrict__ x,
                                                int ldn, int n, int kb, int khalf) {
  const float* p = x + (size_t)(kb + khalf) * ldn + n;
  v16bf b;
#pragma unroll
  for (int e = 0; e < 16; ++e) b[e] = (bf16_t)p[(size_t)e * ldn];
  return b;
}

#define WMMA_BF16(a, b, c) \
  __builtin_amdgcn_wmma_f32_16x16x32_bf16(false, (a), false, (b), (short)0, (c), false, false)

// ---------------------------------------------------------------------------
// Kernels
// ---------------------------------------------------------------------------

__global__ void k_f32_to_bf16(const float* __restrict__ src,
                              bf16_t* __restrict__ dst, int n) {
  int i = blockIdx.x * blockDim.x + threadIdx.x;
  if (i < n) dst[i] = (bf16_t)src[i];
}

// XW[(b*T+t), :] = bf16(feats[data[b,t]]) @ w_ih^T   (M=B*T, N=512, K=128)
// grid = ((B*T)/16, 4), block = 32 (one wave per 16x128 slab)
__global__ void k_xw_gemm(const float* __restrict__ feats,
                          const int* __restrict__ data,
                          const bf16_t* __restrict__ wih,  // (512,128) bf16
                          float* __restrict__ XW, int T) {
  const int lane = threadIdx.x & 31;
  const int l16 = lane & 15, half = lane >> 4, khalf = half << 4, mo = half * 8;
  const int mtile = blockIdx.x, ngrp = blockIdx.y;
  const int m = mtile * 16 + l16;
  const int b = m / T, t = m % T;
  const float* arow = feats + (size_t)data[b * L_N + t] * F_N;

  v8f acc[8] = {};
  for (int kb = 0; kb < F_N; kb += 32) {
    v16bf a = a_frag_f32(arow, kb, half);
#pragma unroll
    for (int nt = 0; nt < 8; ++nt) {
      int n = (ngrp * 8 + nt) * 16 + l16;
      v16bf bb = b_frag_bf16(wih, F_N, n, kb, khalf);
      acc[nt] = WMMA_BF16(a, bb, acc[nt]);
    }
  }
#pragma unroll
  for (int nt = 0; nt < 8; ++nt) {
    int n = (ngrp * 8 + nt) * 16 + l16;
#pragma unroll
    for (int j = 0; j < 8; ++j)
      XW[(size_t)(mtile * 16 + mo + j) * G4H + n] = acc[nt][j];
  }
}

// wf = W(AxA, f32) @ X(Ax128, f32), bf16 WMMA.  grid=(A/16, 128/16), block=32
__global__ void k_wf_gemm(const float* __restrict__ W,
                          const float* __restrict__ X,
                          float* __restrict__ wf) {
  const int lane = threadIdx.x & 31;
  const int l16 = lane & 15, half = lane >> 4, khalf = half << 4, mo = half * 8;
  const float* arow = W + (size_t)(blockIdx.x * 16 + l16) * A_N;
  const int n = blockIdx.y * 16 + l16;

  v8f acc = {};
  for (int kb = 0; kb < A_N; kb += 32) {
    __builtin_prefetch(arow + kb + 128, 0, 0);  // global_prefetch_b8
    v16bf a = a_frag_f32(arow, kb, half);
    v16bf b = b_frag_f32_kxn(X, F_N, n, kb, khalf);
    acc = WMMA_BF16(a, b, acc);
  }
#pragma unroll
  for (int j = 0; j < 8; ++j)
    wf[(size_t)(blockIdx.x * 16 + mo + j) * F_N + n] = acc[j];
}

// Sequential LSTM part: gates = XW[:,t,:] + h @ w_hh^T (+biases), then
// elementwise cell update; final h projected by out_w^T + out_b.
// grid = B/16 workgroups (16 batch rows each), block = 256 (8 waves).
__global__ __launch_bounds__(256) void k_lstm_rec(
    const float* __restrict__ XW, const bf16_t* __restrict__ whh,
    const bf16_t* __restrict__ outw, const float* __restrict__ b_ih,
    const float* __restrict__ b_hh, const float* __restrict__ out_b,
    float* __restrict__ out, int T) {
  __shared__ float gatesS[16 * G4H];   // 32 KB
  __shared__ float cS[16 * H_N];       //  8 KB
  __shared__ bf16_t hS[16 * H_N];      //  4 KB
  const int tid = threadIdx.x;
  const int lane = tid & 31, wid = tid >> 5;
  const int l16 = lane & 15, half = lane >> 4, khalf = half << 4, mo = half * 8;
  const int bbase = blockIdx.x * 16;

  for (int i = tid; i < 16 * H_N; i += 256) { cS[i] = 0.f; hS[i] = (bf16_t)0.f; }
  __syncthreads();

  for (int t = 0; t < T; ++t) {
    // --- gates tile (16 x 512): init from precomputed XW, add h @ w_hh^T ---
    v8f acc[4];
#pragma unroll
    for (int q = 0; q < 4; ++q) {
      int n = (wid * 4 + q) * 16 + l16;
      v8f c;
#pragma unroll
      for (int j = 0; j < 8; ++j)
        c[j] = XW[((size_t)(bbase + mo + j) * T + t) * G4H + n];
      acc[q] = c;
    }
#pragma unroll
    for (int kb = 0; kb < H_N; kb += 32) {
      v16bf a = a_frag_bf16(hS + l16 * H_N, kb, half);
#pragma unroll
      for (int q = 0; q < 4; ++q) {
        int n = (wid * 4 + q) * 16 + l16;
        v16bf bb = b_frag_bf16(whh, H_N, n, kb, khalf);
        acc[q] = WMMA_BF16(a, bb, acc[q]);
      }
    }
#pragma unroll
    for (int q = 0; q < 4; ++q) {
      int n = (wid * 4 + q) * 16 + l16;
#pragma unroll
      for (int j = 0; j < 8; ++j) gatesS[(mo + j) * G4H + n] = acc[q][j];
    }
    __syncthreads();
    // --- elementwise cell update (2048 elems / 256 threads) ---
#pragma unroll
    for (int k = 0; k < 8; ++k) {
      int idx = tid * 8 + k;
      int m = idx >> 7, hi = idx & 127;
      float gi = gatesS[m * G4H + hi]       + b_ih[hi]       + b_hh[hi];
      float gf = gatesS[m * G4H + 128 + hi] + b_ih[128 + hi] + b_hh[128 + hi];
      float gg = gatesS[m * G4H + 256 + hi] + b_ih[256 + hi] + b_hh[256 + hi];
      float go = gatesS[m * G4H + 384 + hi] + b_ih[384 + hi] + b_hh[384 + hi];
      float ig = 1.f / (1.f + __expf(-gi));
      float fg = 1.f / (1.f + __expf(-gf));
      float g  = tanhf(gg);
      float og = 1.f / (1.f + __expf(-go));
      float c  = fg * cS[m * H_N + hi] + ig * g;
      cS[m * H_N + hi] = c;
      hS[m * H_N + hi] = (bf16_t)(og * tanhf(c));
    }
    __syncthreads();
  }

  // --- output projection: out = h @ out_w^T + out_b (8 waves x 8 ntiles) ---
  {
    int n = wid * 16 + l16;
    v8f acc = {};
#pragma unroll
    for (int kb = 0; kb < H_N; kb += 32) {
      v16bf a = a_frag_bf16(hS + l16 * H_N, kb, half);
      v16bf bb = b_frag_bf16(outw, H_N, n, kb, khalf);
      acc = WMMA_BF16(a, bb, acc);
    }
    float ob = out_b[n];
#pragma unroll
    for (int j = 0; j < 8; ++j)
      out[(size_t)(bbase + mo + j) * F_N + n] = acc[j] + ob;
  }
}

// Per-edge attention score + row sums.  grid = B/8, block = 256 (wave per edge)
__global__ void k_scores(const float* __restrict__ X, const float* __restrict__ caseF,
                         const int* __restrict__ data, float* __restrict__ s,
                         float* __restrict__ rowsum) {
  int wid = threadIdx.x >> 5, lane = threadIdx.x & 31;
  int b = blockIdx.x * 8 + wid;
  int cur = data[b * L_N + (L_N - 2)];
  int tgt = data[b * L_N + (L_N - 1)];
  float acc = 0.f;
  for (int f = lane; f < F_N; f += 32) {
    float d = X[(size_t)cur * F_N + f] + caseF[(size_t)b * F_N + f]
            - X[(size_t)tgt * F_N + f];
    acc += d * d;
  }
#pragma unroll
  for (int off = 16; off > 0; off >>= 1) acc += __shfl_down(acc, off, 32);
  if (lane == 0) {
    float sc = __expf(-sqrtf(acc));
    s[b] = sc;
    atomicAdd(&rowsum[tgt], sc);
  }
}

// out[tgt] += (s/rowsum) * wf[cur]  (out pre-initialized to wf)
__global__ void k_scatter(const float* __restrict__ wf, const float* __restrict__ s,
                          const float* __restrict__ rowsum, const int* __restrict__ data,
                          float* __restrict__ out) {
  int g = blockIdx.x * blockDim.x + threadIdx.x;  // B*F
  int b = g >> 7, f = g & 127;
  int cur = data[b * L_N + (L_N - 2)];
  int tgt = data[b * L_N + (L_N - 1)];
  float sn = s[b] / (rowsum[tgt] + EPSV);
  atomicAdd(&out[(size_t)tgt * F_N + f], sn * wf[(size_t)cur * F_N + f]);
}

__global__ void k_relu(float* __restrict__ x, int n) {
  int i = blockIdx.x * blockDim.x + threadIdx.x;
  if (i < n) x[i] = fmaxf(x[i], 0.f);
}

__global__ void k_pre(const float* __restrict__ conv, const float* __restrict__ prerel,
                      const int* __restrict__ data, float* __restrict__ pre) {
  int g = blockIdx.x * blockDim.x + threadIdx.x;  // B*F
  int b = g >> 7, f = g & 127;
  int tgt = data[b * L_N + (L_N - 1)];
  pre[g] = conv[(size_t)tgt * F_N + f] + prerel[g];
}

// squared L2 norm of each 128-wide row; one wave per row. grid = rows/8
__global__ void k_rownorm2(const float* __restrict__ X, float* __restrict__ out,
                           int rows) {
  int wid = threadIdx.x >> 5, lane = threadIdx.x & 31;
  int row = blockIdx.x * 8 + wid;
  if (row >= rows) return;
  float acc = 0.f;
  for (int f = lane; f < F_N; f += 32) {
    float v = X[(size_t)row * F_N + f];
    acc += v * v;
  }
#pragma unroll
  for (int off = 16; off > 0; off >>= 1) acc += __shfl_down(acc, off, 32);
  if (lane == 0) out[row] = acc;
}

// logits[b,a] = -sqrt(|pre_b|^2 + |emb_a|^2 - 2*pre.emb) using exact-f32 WMMA
// grid = (B/16, A/16), block = 32
__global__ void k_logits(const float* __restrict__ pre, const float* __restrict__ emb,
                         const float* __restrict__ pn2, const float* __restrict__ en2,
                         float* __restrict__ outL) {
  const int lane = threadIdx.x & 31;
  const int l16 = lane & 15, half = lane >> 4, mo = half * 8;
  const float* prow = pre + (size_t)(blockIdx.x * 16 + l16) * F_N + half * 2;
  const float* erow = emb + (size_t)(blockIdx.y * 16 + l16) * F_N + half * 2;
  v8f acc = {};
  for (int kb = 0; kb < F_N; kb += 4) {
    v2f a, b;
    a[0] = prow[kb]; a[1] = prow[kb + 1];
    b[0] = erow[kb]; b[1] = erow[kb + 1];
    acc = __builtin_amdgcn_wmma_f32_16x16x4_f32(false, a, false, b, (short)0, acc,
                                                false, false);
  }
  const int an = blockIdx.y * 16 + l16;
  float e2 = en2[an];
#pragma unroll
  for (int j = 0; j < 8; ++j) {
    int bm = blockIdx.x * 16 + mo + j;
    float d2 = pn2[bm] + e2 - 2.f * acc[j];
    outL[(size_t)bm * A_N + an] = -sqrtf(fmaxf(d2, 0.f));
  }
}

// ---------------------------------------------------------------------------
// Host launcher
// ---------------------------------------------------------------------------
extern "C" void kernel_launch(void* const* d_in, const int* in_sizes, int n_in,
                              void* d_out, int out_size, void* d_ws, size_t ws_size,
                              hipStream_t stream) {
  const int*   data  = (const int*)d_in[0];
  const float* emb   = (const float*)d_in[1];
  const float* Watt  = (const float*)d_in[2];
  const float* w_ih  = (const float*)d_in[3];
  const float* w_hh  = (const float*)d_in[4];
  const float* b_ih  = (const float*)d_in[5];
  const float* b_hh  = (const float*)d_in[6];
  const float* out_w = (const float*)d_in[7];
  const float* out_b = (const float*)d_in[8];
  float* logits = (float*)d_out;

  char* ws = (char*)d_ws;
  size_t off = 0;
  auto alloc = [&](size_t bytes) -> char* {
    char* p = ws + off;
    off += (bytes + 255) & ~(size_t)255;
    return p;
  };
  float*  XW     = (float*)alloc((size_t)B_N * 63 * G4H * 4);
  float*  wf     = (float*)alloc((size_t)A_N * F_N * 4);
  float*  x1     = (float*)alloc((size_t)A_N * F_N * 4);
  float*  x2     = (float*)alloc((size_t)A_N * F_N * 4);
  float*  caseF  = (float*)alloc((size_t)B_N * F_N * 4);
  float*  prerel = (float*)alloc((size_t)B_N * F_N * 4);
  float*  pre    = (float*)alloc((size_t)B_N * F_N * 4);
  float*  sbuf   = (float*)alloc(B_N * 4);
  float*  rowsum = (float*)alloc(A_N * 4);
  float*  pn2    = (float*)alloc(B_N * 4);
  float*  en2    = (float*)alloc(A_N * 4);
  bf16_t* wihB   = (bf16_t*)alloc((size_t)G4H * F_N * 2);
  bf16_t* whhB   = (bf16_t*)alloc((size_t)G4H * H_N * 2);
  bf16_t* outwB  = (bf16_t*)alloc((size_t)F_N * H_N * 2);

  // one-time weight conversions (cheap, deterministic every call)
  k_f32_to_bf16<<<(G4H * F_N + 255) / 256, 256, 0, stream>>>(w_ih, wihB, G4H * F_N);
  k_f32_to_bf16<<<(G4H * H_N + 255) / 256, 256, 0, stream>>>(w_hh, whhB, G4H * H_N);
  k_f32_to_bf16<<<(F_N * H_N + 255) / 256, 256, 0, stream>>>(out_w, outwB, F_N * H_N);

  // LSTM #1 on emb[data[:, :63]] -> case_feats
  {
    int T = 63;
    dim3 g((B_N * T) / 16, 4);
    k_xw_gemm<<<g, 32, 0, stream>>>(emb, data, wihB, XW, T);
    k_lstm_rec<<<B_N / 16, 256, 0, stream>>>(XW, whhB, outwB, b_ih, b_hh, out_b,
                                             caseF, T);
  }

  // GAT layers
  const float* xin = emb;
  float* xout = x1;
  for (int l = 0; l < 2; ++l) {
    const float* W = Watt + (size_t)l * A_N * A_N;
    dim3 g(A_N / 16, F_N / 16);
    k_wf_gemm<<<g, 32, 0, stream>>>(W, xin, wf);
    hipMemsetAsync(rowsum, 0, A_N * 4, stream);
    k_scores<<<B_N / 8, 256, 0, stream>>>(xin, caseF, data, sbuf, rowsum);
    hipMemcpyAsync(xout, wf, (size_t)A_N * F_N * 4, hipMemcpyDeviceToDevice, stream);
    k_scatter<<<(B_N * F_N) / 256, 256, 0, stream>>>(wf, sbuf, rowsum, data, xout);
    k_relu<<<(A_N * F_N) / 256, 256, 0, stream>>>(xout, A_N * F_N);
    xin = xout;
    xout = x2;
  }
  const float* conv = xin;  // == x2

  // LSTM #2 on conv_feat[data[:, :62]] -> pre_rel
  {
    int T = 62;
    dim3 g((B_N * T) / 16, 4);
    k_xw_gemm<<<g, 32, 0, stream>>>(conv, data, wihB, XW, T);
    k_lstm_rec<<<B_N / 16, 256, 0, stream>>>(XW, whhB, outwB, b_ih, b_hh, out_b,
                                             prerel, T);
  }

  // pre = conv_feat[tgt] + pre_rel ; distance head
  k_pre<<<(B_N * F_N) / 256, 256, 0, stream>>>(conv, prerel, data, pre);
  k_rownorm2<<<B_N / 8, 256, 0, stream>>>(pre, pn2, B_N);
  k_rownorm2<<<A_N / 8, 256, 0, stream>>>(emb, en2, A_N);
  k_logits<<<dim3(B_N / 16, A_N / 16), 32, 0, stream>>>(pre, emb, pn2, en2, logits);
}
